// MHA_31714038513852
// MI455X (gfx1250) — matrix operand order repaired and checked by reading, not compiled, not run on hardware
//
#include <hip/hip_runtime.h>

// MHA forward for MI455X (gfx1250): bf16 WMMA everywhere, f32 accumulate.
// Pipeline: cvt(fp32->bf16) -> QKV GEMMs (V transposed) -> RoPE(Q,K)
//           -> flash attention (WMMA + LDS P-transpose) -> out GEMM (fp32 out).
// GEMM stages weight tiles block-wide into LDS with CDNA5 async global->LDS
// DMA (ASYNCcnt), double-buffered so the DMA overlaps the WMMA pipeline.

#define B_  2
#define S_  2048
#define D_  2048
#define H_  16
#define HD_ 128

typedef __attribute__((ext_vector_type(16))) __bf16        v16bf;
typedef __attribute__((ext_vector_type(8)))  float         v8f;
typedef __attribute__((ext_vector_type(4)))  unsigned int  u32x4;
typedef __attribute__((ext_vector_type(8)))  unsigned int  u32x8;

__device__ __forceinline__ unsigned short f2bf(float f) {
  unsigned int u = __float_as_uint(f);
  return (unsigned short)((u + 0x7FFFu + ((u >> 16) & 1u)) >> 16);  // RNE
}
__device__ __forceinline__ float bf2f(unsigned short h) {
  return __uint_as_float(((unsigned int)h) << 16);
}
// Assemble a 16x32 bf16 A-fragment lane slice from two adjacent 16B runs.
__device__ __forceinline__ v16bf frag_join(u32x4 lo, u32x4 hi) {
  u32x8 a;
  a[0] = lo[0]; a[1] = lo[1]; a[2] = lo[2]; a[3] = lo[3];
  a[4] = hi[0]; a[5] = hi[1]; a[6] = hi[2]; a[7] = hi[3];
  return __builtin_bit_cast(v16bf, a);
}
// 32x16 bf16 B-fragment lane slice: one contiguous 32B run.
__device__ __forceinline__ v16bf frag_ld32(const unsigned short* p) {
  return __builtin_bit_cast(v16bf, *(const u32x8*)p);
}
__device__ __forceinline__ v8f wmma_bf16(v16bf a, v16bf b, v8f c) {
  return __builtin_amdgcn_wmma_f32_16x16x32_bf16(false, a, false, b,
                                                 (short)0, c, false, false);
}

// ---- CDNA5 async global->LDS DMA (ASYNCcnt) --------------------------------
// global_load_async_to_lds_b128: VDST = LDS byte offset (generic LDS addr low
// 32 bits), VADDR = 64-bit global address. One instruction moves 16B/lane.
__device__ __forceinline__ void async_ld_b128(const void* g, void* l) {
  unsigned           loff = (unsigned)(unsigned long long)(__SIZE_TYPE__)l;
  unsigned long long ga   = (unsigned long long)(__SIZE_TYPE__)g;
  asm volatile("global_load_async_to_lds_b128 %0, %1, off"
               :: "v"(loff), "v"(ga) : "memory");
}
__device__ __forceinline__ void wait_async0() {
#if __has_builtin(__builtin_amdgcn_s_wait_asynccnt)
  __builtin_amdgcn_s_wait_asynccnt(0);
#else
  asm volatile("s_wait_asynccnt 0x0" ::: "memory");
#endif
}

// ---------------------------------------------------------------- convert
__global__ void k_cvt(const float* __restrict__ in,
                      unsigned short* __restrict__ out, int n) {
  for (int i = blockIdx.x * blockDim.x + threadIdx.x; i < n;
       i += blockDim.x * gridDim.x)
    out[i] = f2bf(in[i]);
}

// ------------------------------------------------------------------- GEMM
// C[row][col] = sum_k A[row][k] * W[col][k]   (A: MxK bf16, W: NxK bf16)
// One wave computes a 16x64 tile; block = 8 waves = 128 rows x 64 cols.
// The 64x32 weight k-slice is shared by all 8 waves -> staged block-wide in
// LDS via async DMA, double-buffered (DMA for k+32 overlaps WMMA on k).
// mode 0: bf16 row-major out; mode 1: bf16 transposed out [(b*N+col)*S_+s];
// mode 2: fp32 row-major out.
#define WT_STRIDE 40   // padded LDS row stride (ushorts): 80B, de-banks reads

__global__ void k_gemm(const unsigned short* __restrict__ A,
                       const unsigned short* __restrict__ W,
                       unsigned short* __restrict__ Cb,
                       float* __restrict__ Cf,
                       int N, int K, int mode) {
  __shared__ __align__(16) unsigned short wtile[2][64 * WT_STRIDE];

  const int lane = threadIdx.x & 31;
  const int wave = threadIdx.x >> 5;
  const int hi   = lane >> 4;        // which 16-lane half
  const int ln   = lane & 15;

  const int row0 = blockIdx.x * 128 + wave * 16;
  const int col0 = blockIdx.y * 64;

  // staging map: thread -> (weight column, 8-element k chunk); 256 thr * 16B
  // = one full 64x32 bf16 k-slice per instruction round.
  const int tcol  = threadIdx.x >> 2;        // 0..63
  const int tkoff = (threadIdx.x & 3) * 8;   // 0,8,16,24
  const unsigned short* wsrc = W + (size_t)(col0 + tcol) * K + tkoff;

  const v8f zero = {0.f, 0.f, 0.f, 0.f, 0.f, 0.f, 0.f, 0.f};
  v8f acc[4];
#pragma unroll
  for (int c = 0; c < 4; ++c) acc[c] = zero;

  const unsigned short* aRow = A + (size_t)(row0 + ln) * K;

  // prologue: stage k-tile 0 into buffer 0
  async_ld_b128(wsrc, &wtile[0][tcol * WT_STRIDE + tkoff]);

  int buf = 0;
  for (int kk = 0; kk < K; kk += 32, buf ^= 1) {
    wait_async0();        // my wave's DMA done
    __syncthreads();      // everyone's DMA visible
    if (kk + 32 < K)      // kick DMA for next k-slice into the other buffer
      async_ld_b128(wsrc + kk + 32,
                    &wtile[buf ^ 1][tcol * WT_STRIDE + tkoff]);

    __builtin_prefetch(aRow + kk + 256, 0, 1);      // global_prefetch_b8
    // A fragment (16x32 bf16): lane holds K=[kk+hi*16,+16)
    u32x4 alo = *(const u32x4*)(aRow + kk + hi * 16);
    u32x4 ahi = *(const u32x4*)(aRow + kk + hi * 16 + 8);
    v16bf av  = frag_join(alo, ahi);

#pragma unroll
    for (int c = 0; c < 4; ++c) {
      // B fragment (32x16): lane = column (ln), K = hi*16..+15, from LDS
      const unsigned short* bp =
          &wtile[buf][(c * 16 + ln) * WT_STRIDE + hi * 16];
      acc[c] = wmma_bf16(av, frag_ld32(bp), acc[c]);
    }
    __syncthreads();      // all waves done reading buf before it is re-staged
  }

  // C layout: VGPR v -> row = row0 + v + 8*hi, col = col0 + c*16 + ln
  if (mode == 0) {
#pragma unroll
    for (int c = 0; c < 4; ++c) {
      const int col = col0 + c * 16 + ln;
#pragma unroll
      for (int v = 0; v < 8; ++v)
        Cb[(size_t)(row0 + v + 8 * hi) * N + col] = f2bf(acc[c][v]);
    }
  } else if (mode == 1) {   // transposed (for V): out[(b*N+col)*S_ + s]
#pragma unroll
    for (int c = 0; c < 4; ++c) {
      const int col = col0 + c * 16 + ln;
#pragma unroll
      for (int v = 0; v < 8; ++v) {
        const int row = row0 + v + 8 * hi;
        const int b   = row >> 11;            // S_ = 2048
        const int s   = row & (S_ - 1);
        Cb[((size_t)(b * N + col)) * S_ + s] = f2bf(acc[c][v]);
      }
    }
  } else {
#pragma unroll
    for (int c = 0; c < 4; ++c) {
      const int col = col0 + c * 16 + ln;
#pragma unroll
      for (int v = 0; v < 8; ++v)
        Cf[(size_t)(row0 + v + 8 * hi) * N + col] = acc[c][v];
    }
  }
}

// ------------------------------------------------------------------- RoPE
// In-place interleaved rotation on bf16 [B*S][D] with per-head dim 128.
__global__ void k_rope(unsigned short* __restrict__ X,
                       const int* __restrict__ pos) {
  const float LOG2_THETA = 13.287712379549449f;   // log2(10000)
  int p = blockIdx.x * blockDim.x + threadIdx.x;
  if (p >= B_ * S_ * D_ / 2) return;
  const int r  = p >> 10;            // D_/2 = 1024 pairs per row
  const int c2 = p & 1023;
  const int i  = c2 & (HD_ / 2 - 1); // pair index within head
  const float pf = (float)pos[r & (S_ - 1)];
  const float inv_freq = exp2f(-((float)(2 * i) / (float)HD_) * LOG2_THETA);
  float sn, cs;
  __sincosf(pf * inv_freq, &sn, &cs);
  unsigned short* q = X + (size_t)r * D_ + 2 * c2;
  const float x1 = bf2f(q[0]), x2 = bf2f(q[1]);
  q[0] = f2bf(x1 * cs - x2 * sn);
  q[1] = f2bf(x1 * sn + x2 * cs);
}

// -------------------------------------------------------- flash attention
// One wave per (b, h, 16-row q tile). 32 keys per iteration (two 16x16 score
// tiles), online softmax, P re-laid out C->A via a same-wave LDS bounce,
// O accumulated with WMMA against transposed V.
__global__ void k_attn(const unsigned short* __restrict__ Q,
                       const unsigned short* __restrict__ Kc,
                       const unsigned short* __restrict__ Vt,
                       unsigned short* __restrict__ O) {
  __shared__ __align__(16) unsigned short pl[8][16 * 32];  // per-wave P tile

  const int lane = threadIdx.x & 31;
  const int wave = threadIdx.x >> 5;
  const int hi   = lane >> 4;
  const int ln   = lane & 15;

  const int tile = blockIdx.x * 8 + wave;     // 4096 tiles total
  const int qt   = tile & (S_ / 16 - 1);
  const int bh   = tile / (S_ / 16);
  const int h    = bh & (H_ - 1);
  const int b    = bh >> 4;

  // Q as four 16x32 A-fragments covering hd = 128.
  const unsigned short* qrow =
      Q + ((size_t)(b * S_ + qt * 16 + ln)) * D_ + h * HD_;
  v16bf qa[4];
#pragma unroll
  for (int t = 0; t < 4; ++t) {
    u32x4 lo = *(const u32x4*)(qrow + t * 32 + hi * 16);
    u32x4 hh = *(const u32x4*)(qrow + t * 32 + hi * 16 + 8);
    qa[t] = frag_join(lo, hh);
  }

  const v8f zero = {0.f, 0.f, 0.f, 0.f, 0.f, 0.f, 0.f, 0.f};
  v8f   o[8];
  float mrow[8], lrow[8];
#pragma unroll
  for (int i = 0; i < 8; ++i) { o[i] = zero; mrow[i] = -1e30f; lrow[i] = 0.f; }

  const float scale = 0.08838834764831845f;   // 1/sqrt(128)
  const int   nkb   = (qt * 16 + 16 + 31) / 32;

  for (int kb = 0; kb < nkb; ++kb) {
    const int s0 = kb * 32;

    // ---- scores: 16 q rows x 32 keys, f32 accumulate
    v8f sc[2];
    sc[0] = zero; sc[1] = zero;
#pragma unroll
    for (int nt = 0; nt < 2; ++nt) {
      const unsigned short* krow =
          Kc + ((size_t)(b * S_ + s0 + nt * 16 + ln)) * D_ + h * HD_ + hi * 16;
      __builtin_prefetch(krow + D_ * 32, 0, 1);
#pragma unroll
      for (int t = 0; t < 4; ++t)
        sc[nt] = wmma_bf16(qa[t], frag_ld32(krow + t * 32), sc[nt]);
    }

    // ---- scale + causal mask + per-row max (reduce across 16-lane half)
    float mloc[8];
#pragma unroll
    for (int v = 0; v < 8; ++v) {
      const int qr = qt * 16 + v + 8 * hi;
      float a0 = sc[0][v] * scale;
      float a1 = sc[1][v] * scale;
      if (s0 + ln      > qr) a0 = -1e30f;
      if (s0 + 16 + ln > qr) a1 = -1e30f;
      sc[0][v] = a0; sc[1][v] = a1;
      float m = fmaxf(a0, a1);
      m = fmaxf(m, __shfl_xor(m, 1, 32));
      m = fmaxf(m, __shfl_xor(m, 2, 32));
      m = fmaxf(m, __shfl_xor(m, 4, 32));
      m = fmaxf(m, __shfl_xor(m, 8, 32));
      mloc[v] = m;
    }

    // ---- online softmax update
#pragma unroll
    for (int v = 0; v < 8; ++v) {
      const float mnew  = fmaxf(mrow[v], mloc[v]);
      const float alpha = __expf(mrow[v] - mnew);
      mrow[v] = mnew;
      const float p0 = __expf(sc[0][v] - mnew);
      const float p1 = __expf(sc[1][v] - mnew);
      sc[0][v] = p0; sc[1][v] = p1;
      float rs = p0 + p1;
      rs += __shfl_xor(rs, 1, 32);
      rs += __shfl_xor(rs, 2, 32);
      rs += __shfl_xor(rs, 4, 32);
      rs += __shfl_xor(rs, 8, 32);
      lrow[v] = lrow[v] * alpha + rs;
#pragma unroll
      for (int n = 0; n < 8; ++n) o[n][v] = o[n][v] * alpha;
    }

    // ---- P: C-layout -> 16x32 bf16 A-fragment via LDS (same-wave DS ops
    // are in-order; compiler barrier stops compile-time reordering).
    unsigned short* pw = &pl[wave][0];
#pragma unroll
    for (int v = 0; v < 8; ++v) {
      const int r = v + 8 * hi;
      pw[r * 32 + ln]      = f2bf(sc[0][v]);
      pw[r * 32 + 16 + ln] = f2bf(sc[1][v]);
    }
    __asm__ volatile("" ::: "memory");
    u32x4 plo = *(const u32x4*)(pw + ln * 32 + hi * 16);
    u32x4 phi = *(const u32x4*)(pw + ln * 32 + hi * 16 + 8);
    const v16bf pav = frag_join(plo, phi);
    __asm__ volatile("" ::: "memory");

    // ---- O += P * V  (V transposed: lane slice is a contiguous 32B run)
#pragma unroll
    for (int n = 0; n < 8; ++n) {
      const unsigned short* vrow =
          Vt + ((size_t)(b * D_ + h * HD_ + n * 16 + ln)) * S_ + s0 + hi * 16;
      o[n] = wmma_bf16(pav, frag_ld32(vrow), o[n]);
    }
  }

  // ---- normalize + store bf16 O
#pragma unroll
  for (int n = 0; n < 8; ++n) {
#pragma unroll
    for (int v = 0; v < 8; ++v) {
      const int q = qt * 16 + v + 8 * hi;
      O[((size_t)(b * S_ + q)) * D_ + h * HD_ + n * 16 + ln] =
          f2bf(o[n][v] / lrow[v]);
    }
  }
}

// ----------------------------------------------------------------- launch
extern "C" void kernel_launch(void* const* d_in, const int* in_sizes, int n_in,
                              void* d_out, int out_size, void* d_ws,
                              size_t ws_size, hipStream_t stream) {
  (void)in_sizes; (void)n_in; (void)out_size; (void)ws_size;
  const float* x  = (const float*)d_in[0];
  const int*   tp = (const int*)d_in[1];
  const float* Wq = (const float*)d_in[2];
  const float* Wk = (const float*)d_in[3];
  const float* Wv = (const float*)d_in[4];
  const float* Wo = (const float*)d_in[5];
  float* out = (float*)d_out;

  const size_t NX = (size_t)B_ * S_ * D_;  // 8,388,608
  const size_t NW = (size_t)D_ * D_;       // 4,194,304
  unsigned short* ws  = (unsigned short*)d_ws;
  unsigned short* xb  = ws;
  unsigned short* Wqb = xb + NX;
  unsigned short* Wkb = Wqb + NW;
  unsigned short* Wvb = Wkb + NW;
  unsigned short* Wob = Wvb + NW;
  unsigned short* Qb  = Wob + NW;
  unsigned short* Kb  = Qb + NX;
  unsigned short* Vt  = Kb + NX;           // transposed: [b][e][s]
  unsigned short* Ob  = Vt + NX;

  k_cvt<<<4096, 256, 0, stream>>>(x,  xb,  (int)NX);
  k_cvt<<<2048, 256, 0, stream>>>(Wq, Wqb, (int)NW);
  k_cvt<<<2048, 256, 0, stream>>>(Wk, Wkb, (int)NW);
  k_cvt<<<2048, 256, 0, stream>>>(Wv, Wvb, (int)NW);
  k_cvt<<<2048, 256, 0, stream>>>(Wo, Wob, (int)NW);

  dim3 gg(32, 32), bb(256);   // 4096 rows / 128, 2048 cols / 64
  k_gemm<<<gg, bb, 0, stream>>>(xb, Wqb, Qb, nullptr, D_, D_, 0);
  k_gemm<<<gg, bb, 0, stream>>>(xb, Wkb, Kb, nullptr, D_, D_, 0);
  k_gemm<<<gg, bb, 0, stream>>>(xb, Wvb, Vt, nullptr, D_, D_, 1);

  const int pairs = B_ * S_ * D_ / 2;
  k_rope<<<(pairs + 255) / 256, 256, 0, stream>>>(Qb, tp);
  k_rope<<<(pairs + 255) / 256, 256, 0, stream>>>(Kb, tp);

  k_attn<<<512, 256, 0, stream>>>(Qb, Kb, Vt, Ob);

  k_gemm<<<gg, bb, 0, stream>>>(Ob, Wob, nullptr, out, D_, D_, 2);
}
